// ConvAttention_81011673137471
// MI455X (gfx1250) — compile-verified, matrix operand order checked by reading, not run on hardware
//
// CDNA5 / gfx1250 fused conv-attention (RAD-TTS aligner style).
// f16 WMMA (v_wmma_f32_16x16x32_f16) for all GEMMs, fp32 accumulate.
#include <hip/hip_runtime.h>
#include <hip/hip_bf16.h>
#include <math.h>

typedef __attribute__((ext_vector_type(16))) _Float16 v16h;
typedef __attribute__((ext_vector_type(8)))  float    v8f;
typedef __attribute__((ext_vector_type(4)))  float    v4f;

#define DEVINL __device__ __forceinline__

// ---------------- problem constants ----------------
constexpr int B_   = 32;
constexpr int T1_  = 1600;
constexpr int T2_  = 400;
constexpr int CK_  = 512;   // n_text
constexpr int CQ_  = 80;    // n_mel
constexpr int O1K_ = 1024;  // n_text*2
constexpr int O1Q_ = 160;   // n_mel*2
constexpr int NA_  = 80;    // n_att
constexpr int PAD_ = 96;    // n_att padded to 3 K-steps of 32

// ---------------- WMMA helpers ----------------
DEVINL v8f zero8() {
  v8f z;
#pragma unroll
  for (int i = 0; i < 8; ++i) z[i] = 0.f;
  return z;
}

DEVINL v16h zerofrag() {
  v16h f;
#pragma unroll
  for (int i = 0; i < 16; ++i) f[i] = (_Float16)0.f;
  return f;
}

DEVINL v16h frag_ld(const _Float16* p) {  // two 16B loads (K runs 0..7 and 16..23)
  v16h f;
  v4f* fp = reinterpret_cast<v4f*>(&f);
  fp[0] = *reinterpret_cast<const v4f*>(p);
  fp[1] = *reinterpret_cast<const v4f*>(p + 16);
  return f;
}

// A/B operand fragment: rows stored K-contiguous with stride ldk (halves, 16B multiple).
// lane&15 selects row, lane bit4 selects the +8 K sub-block (matches 16x16 C/D pairing).
DEVINL v16h frag_rows(const _Float16* base, int ldk, int row0, int kbase, int lane) {
  int r  = row0 + (lane & 15);
  int hb = (lane & 16) ? 8 : 0;
  return frag_ld(base + (size_t)r * ldk + kbase + hb);
}

DEVINL v16h frag_rows_guard(const _Float16* base, int ldk, int row0, int rmax,
                            int kbase, int lane) {
  int r = row0 + (lane & 15);
  if (r < 0 || r >= rmax) return zerofrag();
  int hb = (lane & 16) ? 8 : 0;
  return frag_ld(base + (size_t)r * ldk + kbase + hb);
}

DEVINL v8f wmma(v16h a, v16h b, v8f c) {
  return __builtin_amdgcn_wmma_f32_16x16x32_f16(false, a, false, b, (short)0, c,
                                                false, false);
}

// ---------------- prep: weights -> f16, tap-major K order, padded ----------------
__global__ void prep_w(const float* __restrict__ kw1, const float* __restrict__ qw1,
                       const float* __restrict__ kw2, const float* __restrict__ qw2,
                       const float* __restrict__ qw3,
                       _Float16* __restrict__ kw1T, _Float16* __restrict__ qw1T,
                       _Float16* __restrict__ kw2h, _Float16* __restrict__ qw2h,
                       _Float16* __restrict__ qw3h) {
  long i = (long)blockIdx.x * blockDim.x + threadIdx.x;
  const long n1 = (long)O1K_ * (3 * CK_);   // kw1T [o][d*512+c]
  const long n2 = (long)O1Q_ * (3 * PAD_);  // qw1T [o][d*96+c], c<80 valid
  const long n3 = (long)NA_ * O1K_;         // kw2h
  const long n4 = (long)NA_ * O1Q_;         // qw2h
  const long n5 = (long)NA_ * PAD_;         // qw3h padded
  if (i < n1) {
    long o = i / (3 * CK_), k = i % (3 * CK_), d = k / CK_, c = k % CK_;
    kw1T[i] = (_Float16)kw1[(o * CK_ + c) * 3 + d];
    return;
  }
  i -= n1;
  if (i < n2) {
    long o = i / (3 * PAD_), k = i % (3 * PAD_), d = k / PAD_, c = k % PAD_;
    qw1T[i] = (_Float16)(c < CQ_ ? qw1[(o * CQ_ + c) * 3 + d] : 0.f);
    return;
  }
  i -= n2;
  if (i < n3) { kw2h[i] = (_Float16)kw2[i]; return; }
  i -= n3;
  if (i < n4) { qw2h[i] = (_Float16)qw2[i]; return; }
  i -= n4;
  if (i < n5) {
    long a = i / PAD_, c = i % PAD_;
    qw3h[i] = (_Float16)(c < CQ_ ? qw3[a * CQ_ + c] : 0.f);
  }
}

// ---------------- prep: activations -> f16, [b][t][c] (K-contiguous) ----------------
__global__ void prep_x(const float* __restrict__ keys, const float* __restrict__ queries,
                       _Float16* __restrict__ kT, _Float16* __restrict__ qT) {
  long i = (long)blockIdx.x * blockDim.x + threadIdx.x;
  const long nk = (long)B_ * T2_ * CK_;
  if (i < nk) {
    long b = i / ((long)T2_ * CK_), r = i % ((long)T2_ * CK_);
    long t = r / CK_, c = r % CK_;
    kT[i] = (_Float16)keys[(b * CK_ + c) * T2_ + t];
    return;
  }
  i -= nk;
  const long nq = (long)B_ * T1_ * PAD_;
  if (i < nq) {
    long b = i / ((long)T1_ * PAD_), r = i % ((long)T1_ * PAD_);
    long t = r / PAD_, c = r % PAD_;
    qT[i] = (_Float16)(c < CQ_ ? queries[(b * CQ_ + c) * T1_ + t] : 0.f);
  }
}

// ---------------- fused keys pipeline: conv3(relu) -> conv1 ----------------
// grid (7, 32), block 128 (4 waves, each owns a 16-column slice of a 64-col tile)
__global__ __launch_bounds__(128) void k_keys(
    const _Float16* __restrict__ kT, const _Float16* __restrict__ kw1T,
    const _Float16* __restrict__ kw2h, const float* __restrict__ kb1,
    const float* __restrict__ kb2, _Float16* __restrict__ kf16,
    float* __restrict__ k2) {
  __shared__ __align__(16) _Float16 b2[64 * 72];  // relu(conv1) chunk, [t][o_local]
  const int tid = threadIdx.x, lane = tid & 31, w = tid >> 5;
  const int tl = lane & 15, hb8 = (lane & 16) ? 8 : 0;
  const int b = blockIdx.y, t0 = blockIdx.x * 64, tsub = w * 16;
  const int tcol = t0 + tsub;
  const _Float16* kTb = kT + (size_t)b * T2_ * CK_;

  v8f acc2[5];
#pragma unroll
  for (int i = 0; i < 5; ++i) acc2[i] = zero8();

  for (int oc = 0; oc < 16; ++oc) {  // 1024 out channels in chunks of 64
    const int o0 = oc * 64;
    __builtin_prefetch(kb1 + o0, 0, 1);
    v8f acc1[4];
#pragma unroll
    for (int i = 0; i < 4; ++i) acc1[i] = zero8();

    for (int s = 0; s < 48; ++s) {  // K = 3 taps * 512 ch, 32 per step
      const int d = s >> 4, cb = (s & 15) * 32;
      v16h bf = frag_rows_guard(kTb, CK_, tcol + d - 1, T2_, cb, lane);
#pragma unroll
      for (int mt = 0; mt < 4; ++mt) {
        v16h af = frag_rows(kw1T + (size_t)(o0 + mt * 16) * (3 * CK_), 3 * CK_, 0,
                            s * 32, lane);
        acc1[mt] = wmma(af, bf, acc1[mt]);
      }
    }
    __syncthreads();  // prev-chunk conv2 reads done before rewrite
#pragma unroll
    for (int mt = 0; mt < 4; ++mt)
#pragma unroll
      for (int v = 0; v < 8; ++v) {
        int m = mt * 16 + v + hb8;
        float val = acc1[mt][v] + kb1[o0 + m];
        b2[(tsub + tl) * 72 + m] = (_Float16)(val > 0.f ? val : 0.f);
      }
    __syncthreads();
#pragma unroll
    for (int ks = 0; ks < 2; ++ks) {  // conv2: K chunk of 64
      v16h bf2 = frag_rows(b2, 72, tsub, ks * 32, lane);
#pragma unroll
      for (int at = 0; at < 5; ++at) {
        v16h af2 = frag_rows(kw2h + (size_t)(at * 16) * O1K_, O1K_, 0,
                             o0 + ks * 32, lane);
        acc2[at] = wmma(af2, bf2, acc2[at]);
      }
    }
  }

  const int t = tcol + tl;
  const bool tok = t < T2_;
  _Float16* kfRow = kf16 + ((size_t)b * T2_ + (tok ? t : 0)) * PAD_;
  float sq = 0.f;
#pragma unroll
  for (int at = 0; at < 5; ++at)
#pragma unroll
    for (int v = 0; v < 8; ++v) {
      int a = at * 16 + v + hb8;
      float val = acc2[at][v] + kb2[a];
      sq += val * val;
      if (tok) kfRow[a] = (_Float16)val;
    }
  if (tok) {
#pragma unroll
    for (int j = 0; j < 8; ++j) kfRow[NA_ + hb8 + j] = (_Float16)0.f;  // pad 80..95
  }
  sq += __shfl_xor(sq, 16, 32);
  if (tok && lane < 16) k2[(size_t)b * T2_ + t] = sq;
}

// ---------------- fused queries pipeline: conv3(relu)->conv1(relu)->conv1 ----------------
// grid (25, 32), block 128
__global__ __launch_bounds__(128) void k_queries(
    const _Float16* __restrict__ qT, const _Float16* __restrict__ qw1T,
    const _Float16* __restrict__ qw2h, const _Float16* __restrict__ qw3h,
    const float* __restrict__ qb1, const float* __restrict__ qb2,
    const float* __restrict__ qb3, _Float16* __restrict__ qf16,
    float* __restrict__ q2) {
  __shared__ __align__(16) _Float16 b2[64 * 168];  // relu(conv1) [t][m<160]
  __shared__ __align__(16) _Float16 b3[64 * 104];  // relu(conv2) [t][a<96]
  const int tid = threadIdx.x, lane = tid & 31, w = tid >> 5;
  const int tl = lane & 15, hb8 = (lane & 16) ? 8 : 0;
  const int b = blockIdx.y, t0 = blockIdx.x * 64, tsub = w * 16;
  const int tcol = t0 + tsub;
  const _Float16* qTb = qT + (size_t)b * T1_ * PAD_;

  // conv1: M=160, K=3*96
  v8f acc1[10];
#pragma unroll
  for (int i = 0; i < 10; ++i) acc1[i] = zero8();
  for (int s = 0; s < 9; ++s) {
    const int d = s / 3, cb = (s % 3) * 32;
    v16h bf = frag_rows_guard(qTb, PAD_, tcol + d - 1, T1_, cb, lane);
#pragma unroll
    for (int mt = 0; mt < 10; ++mt) {
      v16h af = frag_rows(qw1T + (size_t)(mt * 16) * (3 * PAD_), 3 * PAD_, 0,
                          s * 32, lane);
      acc1[mt] = wmma(af, bf, acc1[mt]);
    }
  }
#pragma unroll
  for (int mt = 0; mt < 10; ++mt)
#pragma unroll
    for (int v = 0; v < 8; ++v) {
      int m = mt * 16 + v + hb8;
      float val = acc1[mt][v] + qb1[m];
      b2[(tsub + tl) * 168 + m] = (_Float16)(val > 0.f ? val : 0.f);
    }
  __syncthreads();

  // conv2: M=80, K=160
  v8f acc2[5];
#pragma unroll
  for (int i = 0; i < 5; ++i) acc2[i] = zero8();
#pragma unroll
  for (int ks = 0; ks < 5; ++ks) {
    v16h bf2 = frag_rows(b2, 168, tsub, ks * 32, lane);
#pragma unroll
    for (int at = 0; at < 5; ++at) {
      v16h af2 = frag_rows(qw2h + (size_t)(at * 16) * O1Q_, O1Q_, 0, ks * 32, lane);
      acc2[at] = wmma(af2, bf2, acc2[at]);
    }
  }
#pragma unroll
  for (int at = 0; at < 5; ++at)
#pragma unroll
    for (int v = 0; v < 8; ++v) {
      int a = at * 16 + v + hb8;
      float val = acc2[at][v] + qb2[a];
      b3[(tsub + tl) * 104 + a] = (_Float16)(val > 0.f ? val : 0.f);
    }
#pragma unroll
  for (int j = 0; j < 8; ++j)  // zero pad cols 80..95
    b3[(tsub + tl) * 104 + NA_ + hb8 + j] = (_Float16)0.f;
  __syncthreads();

  // conv3: M=80, K=96
  v8f acc3[5];
#pragma unroll
  for (int i = 0; i < 5; ++i) acc3[i] = zero8();
#pragma unroll
  for (int ks = 0; ks < 3; ++ks) {
    v16h bf3 = frag_rows(b3, 104, tsub, ks * 32, lane);
#pragma unroll
    for (int at = 0; at < 5; ++at) {
      v16h af3 = frag_rows(qw3h + (size_t)(at * 16) * PAD_, PAD_, 0, ks * 32, lane);
      acc3[at] = wmma(af3, bf3, acc3[at]);
    }
  }

  const int t = tcol + tl;  // always < 1600
  _Float16* qfRow = qf16 + ((size_t)b * T1_ + t) * PAD_;
  float sq = 0.f;
#pragma unroll
  for (int at = 0; at < 5; ++at)
#pragma unroll
    for (int v = 0; v < 8; ++v) {
      int a = at * 16 + v + hb8;
      float val = acc3[at][v] + qb3[a];
      sq += val * val;
      qfRow[a] = (_Float16)val;
    }
#pragma unroll
  for (int j = 0; j < 8; ++j) qfRow[NA_ + hb8 + j] = (_Float16)0.f;
  sq += __shfl_xor(sq, 16, 32);
  if (lane < 16) q2[(size_t)b * T1_ + t] = sq;
}

// ---------------- attention ----------------
// cross-lane (xor<16) + cross-wave (LDS) row reduction; rows m = v + hb8
DEVINL void xred(float (&val)[8], float (*red)[16], int w, int lane, int hb8,
                 bool ismax) {
#pragma unroll
  for (int off = 1; off < 16; off <<= 1) {
#pragma unroll
    for (int v = 0; v < 8; ++v) {
      float o = __shfl_xor(val[v], off, 32);
      val[v] = ismax ? fmaxf(val[v], o) : (val[v] + o);
    }
  }
  if (lane == 0) {
#pragma unroll
    for (int v = 0; v < 8; ++v) red[w][v] = val[v];
  } else if (lane == 16) {
#pragma unroll
    for (int v = 0; v < 8; ++v) red[w][8 + v] = val[v];
  }
  __syncthreads();
#pragma unroll
  for (int v = 0; v < 8; ++v) {
    float a = red[0][v + hb8];
#pragma unroll
    for (int w2 = 1; w2 < 5; ++w2) {
      float x = red[w2][v + hb8];
      a = ismax ? fmaxf(a, x) : (a + x);
    }
    val[v] = a;
  }
  __syncthreads();
}

// grid (100, 32), block 160 (5 waves x 5 N-tiles = 25 tiles = 400 cols)
__global__ __launch_bounds__(160) void k_attn(
    const _Float16* __restrict__ qf16, const _Float16* __restrict__ kf16,
    const float* __restrict__ q2, const float* __restrict__ k2,
    const float* __restrict__ prior, const unsigned char* __restrict__ mask,
    float* __restrict__ outA, float* __restrict__ outL) {
  __shared__ float red[5][16];
  const int tid = threadIdx.x, lane = tid & 31, w = tid >> 5;
  const int tl = lane & 15, hb8 = (lane & 16) ? 8 : 0;
  const int b = blockIdx.y, t10 = blockIdx.x * 16;
  const float NEG_INF = -__builtin_inff();

  const _Float16* qfb = qf16 + ((size_t)b * T1_ + t10) * PAD_;
  v16h af[3];
#pragma unroll
  for (int ks = 0; ks < 3; ++ks) af[ks] = frag_rows(qfb, PAD_, 0, ks * 32, lane);

  float q2v[8];
#pragma unroll
  for (int v = 0; v < 8; ++v) q2v[v] = q2[(size_t)b * T1_ + t10 + v + hb8];

  float dmat[5][8];
  int t2b[5];
#pragma unroll
  for (int j = 0; j < 5; ++j) {
    t2b[j] = (w * 5 + j) * 16;
    v8f acc = zero8();
    const _Float16* kfb = kf16 + ((size_t)b * T2_ + t2b[j]) * PAD_;
#pragma unroll
    for (int ks = 0; ks < 3; ++ks)
      acc = wmma(af[ks], frag_rows(kfb, PAD_, 0, ks * 32, lane), acc);
    float k2j = k2[(size_t)b * T2_ + t2b[j] + tl];
#pragma unroll
    for (int v = 0; v < 8; ++v)
      dmat[j][v] = -5.0e-4f * (q2v[v] + k2j - 2.f * acc[v]);
  }

  // log-softmax over all 400 cols
  float r8[8];
#pragma unroll
  for (int v = 0; v < 8; ++v) {
    float m = dmat[0][v];
#pragma unroll
    for (int j = 1; j < 5; ++j) m = fmaxf(m, dmat[j][v]);
    r8[v] = m;
  }
  xred(r8, red, w, lane, hb8, true);
  float gmax[8];
#pragma unroll
  for (int v = 0; v < 8; ++v) gmax[v] = r8[v];
#pragma unroll
  for (int v = 0; v < 8; ++v) {
    float s = 0.f;
#pragma unroll
    for (int j = 0; j < 5; ++j) s += __expf(dmat[j][v] - gmax[v]);
    r8[v] = s;
  }
  xred(r8, red, w, lane, hb8, false);
  float logZ[8];
#pragma unroll
  for (int v = 0; v < 8; ++v) logZ[v] = gmax[v] + __logf(r8[v]);

  const float* prow = prior + ((size_t)b * T1_ + t10) * T2_;
  float* outLrow = outL + ((size_t)b * T1_ + t10) * T2_;
  float* outArow = outA + ((size_t)b * T1_ + t10) * T2_;

  float xm[5][8];
#pragma unroll
  for (int j = 0; j < 5; ++j) {
    const int t2 = t2b[j] + tl;
    const int mk = (int)mask[(size_t)b * T2_ + t2];
#pragma unroll
    for (int v = 0; v < 8; ++v) {
      const size_t idx = (size_t)(v + hb8) * T2_ + t2;
      float lp = dmat[j][v] - logZ[v] + __logf(prow[idx] + 1e-8f);
      outLrow[idx] = lp;
      xm[j][v] = mk ? NEG_INF : lp;
    }
  }

  // masked softmax
#pragma unroll
  for (int v = 0; v < 8; ++v) {
    float m = NEG_INF;
#pragma unroll
    for (int j = 0; j < 5; ++j) m = fmaxf(m, xm[j][v]);
    r8[v] = m;
  }
  xred(r8, red, w, lane, hb8, true);
  float gm2[8];
#pragma unroll
  for (int v = 0; v < 8; ++v) gm2[v] = r8[v];
#pragma unroll
  for (int v = 0; v < 8; ++v) {
    float s = 0.f;
#pragma unroll
    for (int j = 0; j < 5; ++j)
      s += (xm[j][v] > NEG_INF) ? __expf(xm[j][v] - gm2[v]) : 0.f;
    r8[v] = s;
  }
  xred(r8, red, w, lane, hb8, false);
#pragma unroll
  for (int j = 0; j < 5; ++j) {
    const int t2 = t2b[j] + tl;
#pragma unroll
    for (int v = 0; v < 8; ++v) {
      float e = (xm[j][v] > NEG_INF) ? __expf(xm[j][v] - gm2[v]) : 0.f;
      float inv = (r8[v] > 0.f) ? 1.f / r8[v] : 0.f;
      outArow[(size_t)(v + hb8) * T2_ + t2] = e * inv;
    }
  }
}

// ---------------- launch ----------------
extern "C" void kernel_launch(void* const* d_in, const int* in_sizes, int n_in,
                              void* d_out, int out_size, void* d_ws, size_t ws_size,
                              hipStream_t stream) {
  (void)in_sizes; (void)n_in; (void)out_size; (void)ws_size;
  const float* queries = (const float*)d_in[0];
  const float* keys    = (const float*)d_in[1];
  const float* prior   = (const float*)d_in[2];
  const unsigned char* mask = (const unsigned char*)d_in[3];
  const float* kw1 = (const float*)d_in[4];
  const float* kb1 = (const float*)d_in[5];
  const float* kw2 = (const float*)d_in[6];
  const float* kb2 = (const float*)d_in[7];
  const float* qw1 = (const float*)d_in[8];
  const float* qb1 = (const float*)d_in[9];
  const float* qw2 = (const float*)d_in[10];
  const float* qb2 = (const float*)d_in[11];
  const float* qw3 = (const float*)d_in[12];
  const float* qb3 = (const float*)d_in[13];

  char* wsp = (char*)d_ws;
  size_t off = 0;
  auto carve = [&](size_t bytes) -> char* {
    char* p = wsp + off;
    off = (off + bytes + 255) & ~(size_t)255;
    return p;
  };
  _Float16* kT   = (_Float16*)carve((size_t)B_ * T2_ * CK_ * 2);
  _Float16* qT   = (_Float16*)carve((size_t)B_ * T1_ * PAD_ * 2);
  _Float16* kw1T = (_Float16*)carve((size_t)O1K_ * 3 * CK_ * 2);
  _Float16* qw1T = (_Float16*)carve((size_t)O1Q_ * 3 * PAD_ * 2);
  _Float16* kw2h = (_Float16*)carve((size_t)NA_ * O1K_ * 2);
  _Float16* qw2h = (_Float16*)carve((size_t)NA_ * O1Q_ * 2);
  _Float16* qw3h = (_Float16*)carve((size_t)NA_ * PAD_ * 2);
  _Float16* kf16 = (_Float16*)carve((size_t)B_ * T2_ * PAD_ * 2);
  _Float16* qf16 = (_Float16*)carve((size_t)B_ * T1_ * PAD_ * 2);
  float* k2 = (float*)carve((size_t)B_ * T2_ * 4);
  float* q2 = (float*)carve((size_t)B_ * T1_ * 4);

  {
    long total = (long)O1K_ * 3 * CK_ + (long)O1Q_ * 3 * PAD_ + (long)NA_ * O1K_ +
                 (long)NA_ * O1Q_ + (long)NA_ * PAD_;
    unsigned grid = (unsigned)((total + 255) / 256);
    prep_w<<<dim3(grid), dim3(256), 0, stream>>>(kw1, qw1, kw2, qw2, qw3, kw1T,
                                                 qw1T, kw2h, qw2h, qw3h);
  }
  {
    long total = (long)B_ * T2_ * CK_ + (long)B_ * T1_ * PAD_;
    unsigned grid = (unsigned)((total + 255) / 256);
    prep_x<<<dim3(grid), dim3(256), 0, stream>>>(keys, queries, kT, qT);
  }
  k_keys<<<dim3(7, 32), dim3(128), 0, stream>>>(kT, kw1T, kw2h, kb1, kb2, kf16, k2);
  k_queries<<<dim3(25, 32), dim3(128), 0, stream>>>(qT, qw1T, qw2h, qw3h, qb1, qb2,
                                                    qb3, qf16, q2);
  float* outA = (float*)d_out;
  float* outL = outA + (size_t)B_ * T1_ * T2_;
  k_attn<<<dim3(100, 32), dim3(160), 0, stream>>>(qf16, kf16, q2, k2, prior, mask,
                                                  outA, outL);
}
